// seq2seqModel_23021024706924
// MI455X (gfx1250) — matrix operand order repaired and checked by reading, not compiled, not run on hardware
//
#include <hip/hip_runtime.h>
#include <hip/hip_bf16.h>

// ---------------------------------------------------------------------------
// Seq2seq (biLSTM encoder + attention decoder) for MI455X / gfx1250, wave32.
//  - All LSTM/fc GEMMs: v_wmma_f32_16x16x32_bf16 (B=16 rows == one WMMA M-tile)
//  - Encoder: weight B-fragments register-resident across the 256-step scan.
//  - Decoder: 16-block cooperative kernel, agent-scope atomic barrier,
//    batch-split f32 attention (block-local) + gate-tile-split WMMA LSTM.
//  - fc GEMM hoisted out of the scan: [4096,256]x[256,32000]; B frags
//    register-resident over 32 M-tiles; A tiles staged into LDS with
//    global_load_async_to_lds_b128 double-buffering (ASYNCcnt).
//  - HBM bound term: 524MB f32 output write ~22us @ 23.3 TB/s; fcW (16MB bf16)
//    stays L2-resident (192MB L2) and is read only 8x total.
// ---------------------------------------------------------------------------

typedef __attribute__((ext_vector_type(16))) __bf16 v16bf;
typedef __attribute__((ext_vector_type(8)))  float  v8f;
typedef unsigned short u16b;

union BFrag { u16b u[16]; v16bf v; };

__device__ __forceinline__ u16b f2bf(float f) {
  union { float f; unsigned u; } x; x.f = f;
  unsigned r = x.u + 0x7FFFu + ((x.u >> 16) & 1u);   // round-to-nearest-even
  return (u16b)(r >> 16);
}
__device__ __forceinline__ float sigmoidf_(float x) { return 1.f / (1.f + expf(-x)); }

// A fragment (16xK slice, 16-bit): lane l -> row l&15, K-chunk
// [kb+kh..+15], kh = 16*(l>=16).
__device__ __forceinline__ v16bf load_a(const u16b* p, int ld, int kbase, int lane) {
  BFrag r;
  int row = lane & 15;
  int kh  = (lane >> 4) << 4;
  const u16b* s = p + row * ld + kbase + kh;
#pragma unroll
  for (int j = 0; j < 16; ++j) r.u[j] = s[j];
  return r.v;
}

// B fragment for gates = X @ W.T: B[k,n] = W[n,k], W row-major [N][K].
__device__ __forceinline__ v16bf load_b(const u16b* w, int ldk, int n0, int kbase, int lane) {
  BFrag r;
  int col = lane & 15;
  int ks  = (lane >> 4) << 4;
  const u16b* s = w + (size_t)(n0 + col) * ldk + kbase + ks;
#pragma unroll
  for (int j = 0; j < 16; ++j) r.u[j] = s[j];
  return r.v;
}

__device__ __forceinline__ v8f wmma_bf16(v16bf a, v16bf b, v8f c) {
  return __builtin_amdgcn_wmma_f32_16x16x32_bf16(false, a, false, b, (short)0, c, false, false);
}

__device__ __forceinline__ v8f splat8(float x) {
  v8f v;
#pragma unroll
  for (int r = 0; r < 8; ++r) v[r] = x;
  return v;
}
__device__ __forceinline__ v8f bias_splat(const float* b, int n0, int lane) {
  return splat8(b[n0 + (lane & 15)]);
}

// Agent-scope grid barrier (monotonic counter, zeroed by k_zero each launch).
__device__ __forceinline__ void gbar(unsigned* cnt, unsigned expected) {
  __threadfence();
  __syncthreads();
  if (threadIdx.x == 0) {
    __hip_atomic_fetch_add(cnt, 1u, __ATOMIC_RELEASE, __HIP_MEMORY_SCOPE_AGENT);
    unsigned v;
    do {
      __builtin_amdgcn_s_sleep(2);
      v = __hip_atomic_load(cnt, __ATOMIC_ACQUIRE, __HIP_MEMORY_SCOPE_AGENT);
    } while (v < expected);
  }
  __syncthreads();
  __threadfence();
}

// Async DMA: stage one 16x256 bf16 A tile (8KB) into LDS. 256 threads x 32B.
__device__ __forceinline__ void stage_a(u16b* dstLds, const u16b* src, int tid) {
  unsigned lds0 = (unsigned)(unsigned long long)dstLds + (unsigned)(tid * 32);
  unsigned off  = (unsigned)(tid * 32);
  unsigned long long base = (unsigned long long)src;
  asm volatile("global_load_async_to_lds_b128 %0, %1, %2"
               :: "v"(lds0), "v"(off), "s"(base) : "memory");
  asm volatile("global_load_async_to_lds_b128 %0, %1, %2 offset:16"
               :: "v"(lds0 + 16), "v"(off), "s"(base) : "memory");
}
__device__ __forceinline__ void wait_async0() {
  asm volatile("s_wait_asynccnt 0" ::: "memory");
}

// ---------------------------------------------------------------------------
__global__ __launch_bounds__(256) void k_zero(unsigned* p) {
  if (threadIdx.x == 0 && blockIdx.x == 0) *p = 0u;
}

__global__ __launch_bounds__(256) void k_f32_to_bf16(const float* __restrict__ src,
                                                     u16b* __restrict__ dst, int n) {
  for (int i = blockIdx.x * 256 + threadIdx.x; i < n; i += gridDim.x * 256)
    dst[i] = f2bf(src[i]);
}

// embeddings: bf16 time-major (encoder WMMA A) and f32 time-major (decoder VALU)
__global__ __launch_bounds__(256) void k_embed_bf(const int* __restrict__ tok,
                                                  const float* __restrict__ emb,
                                                  u16b* __restrict__ out, int seqLen) {
  int total = seqLen * 16 * 64;
  for (int i = blockIdx.x * 256 + threadIdx.x; i < total; i += gridDim.x * 256) {
    int e = i & 63, b = (i >> 6) & 15, s = i >> 10;
    out[i] = f2bf(emb[(size_t)tok[b * seqLen + s] * 64 + e]);
  }
}
__global__ __launch_bounds__(256) void k_embed_f32(const int* __restrict__ tok,
                                                   const float* __restrict__ emb,
                                                   float* __restrict__ out, int seqLen) {
  int total = seqLen * 16 * 64;
  for (int i = blockIdx.x * 256 + threadIdx.x; i < total; i += gridDim.x * 256) {
    int e = i & 63, b = (i >> 6) & 15, s = i >> 10;
    out[i] = emb[(size_t)tok[b * seqLen + s] * 64 + e];
  }
}

// ---------------------------------------------------------------------------
// Encoder: 2 blocks (fwd/bwd), 8 waves. Wave w owns gate tiles {w,8+w,16+w,24+w}
// (i,f,g,o aligned) so the cell update runs in-register on the D fragments.
// Weight B fragments preloaded once (loop-invariant across 256 steps).
// ---------------------------------------------------------------------------
__global__ __launch_bounds__(256) void k_encoder(
    const u16b* __restrict__ srcEmb,   // [S][16][64] bf16
    const u16b* __restrict__ WihF, const u16b* __restrict__ WhhF, const float* __restrict__ bF,
    const u16b* __restrict__ WihB, const u16b* __restrict__ WhhB, const float* __restrict__ bB,
    float* __restrict__ encF,          // [16][S][256] f32
    float* __restrict__ h0, float* __restrict__ c0)  // [16][256]
{
  const int S = 256;
  int dir = blockIdx.x;
  const u16b* Wih  = dir ? WihB : WihF;
  const u16b* Whh  = dir ? WhhB : WhhF;
  const float* bia = dir ? bB   : bF;

  __shared__ u16b  hbf[16 * 128];
  __shared__ float cS [16 * 128];

  int tid = threadIdx.x, lane = tid & 31, w = tid >> 5;
  for (int i = tid; i < 16 * 128; i += 256) { hbf[i] = 0; cS[i] = 0.f; }
  __syncthreads();

  const int tiles[4] = { w, 8 + w, 16 + w, 24 + w };   // i,f,g,o
  v16bf bih[4][2], bhh[4][4];
  v8f   bsp[4];
#pragma unroll
  for (int p = 0; p < 4; ++p) {
    int n0 = tiles[p] * 16;
#pragma unroll
    for (int kk = 0; kk < 2; ++kk) bih[p][kk] = load_b(Wih, 64, n0, kk * 32, lane);
#pragma unroll
    for (int kk = 0; kk < 4; ++kk) bhh[p][kk] = load_b(Whh, 128, n0, kk * 32, lane);
    bsp[p] = bias_splat(bia, n0, lane);
  }

  int nH = w * 16 + (lane & 15);
  int mb = (lane >> 4) << 3;

  for (int t = 0; t < S; ++t) {
    int s = dir ? (S - 1 - t) : t;
    const u16b* x = srcEmb + (size_t)s * (16 * 64);
    v16bf xa[2], ha[4];
#pragma unroll
    for (int kk = 0; kk < 2; ++kk) xa[kk] = load_a(x, 64, kk * 32, lane);
#pragma unroll
    for (int kk = 0; kk < 4; ++kk) ha[kk] = load_a(hbf, 128, kk * 32, lane);

    v8f acc[4];
#pragma unroll
    for (int p = 0; p < 4; ++p) {
      acc[p] = bsp[p];
#pragma unroll
      for (int kk = 0; kk < 2; ++kk) acc[p] = wmma_bf16(xa[kk], bih[p][kk], acc[p]);
#pragma unroll
      for (int kk = 0; kk < 4; ++kk) acc[p] = wmma_bf16(ha[kk], bhh[p][kk], acc[p]);
    }
    __syncthreads();   // all waves done reading hbf
#pragma unroll
    for (int r = 0; r < 8; ++r) {
      int m = mb + r;
      float iv = sigmoidf_(acc[0][r]);
      float fv = sigmoidf_(acc[1][r]);
      float gv = tanhf(acc[2][r]);
      float ov = sigmoidf_(acc[3][r]);
      float cN = fv * cS[m * 128 + nH] + iv * gv;
      float hN = ov * tanhf(cN);
      cS[m * 128 + nH] = cN;
      hbf[m * 128 + nH] = f2bf(hN);
      encF[((size_t)(m * S) + s) * 256 + dir * 128 + nH] = hN;
      if (t == S - 1) {
        h0[m * 256 + dir * 128 + nH] = hN;
        c0[m * 256 + dir * 128 + nH] = cN;
      }
    }
    __syncthreads();
  }
}

// ---------------------------------------------------------------------------
// Decoder: 16 cooperative blocks, 2 agent barriers per step.
// Phase A (block = batch b): scores -> softmax -> context -> attn proj, all
//   block-local f32 VALU (matvec-shaped; WMMA would waste 15/16 lanes).
// Phase C (block = gate col-tile k): WMMA LSTM gates (4 waves, i/f/g/o),
//   cell update for cols [16k,16k+16), publish h (f32 + bf16) and hAll.
// ---------------------------------------------------------------------------
__global__ __launch_bounds__(256) void k_decoder(
    const float* __restrict__ tgtE,    // [T][16][64] f32
    const float* __restrict__ encF,    // [16][S][256] f32
    const float* __restrict__ h0, const float* __restrict__ c0,
    const float* __restrict__ attnW,   // [256][320] f32
    const float* __restrict__ attnB,
    const u16b* __restrict__ decWih,   // [1024][256] bf16
    const u16b* __restrict__ decWhh,
    const float* __restrict__ decB,
    float* __restrict__ hF,            // [16][256] f32 (global state)
    u16b* __restrict__ hbf,            // [16][256] bf16
    u16b* __restrict__ xbf,            // [16][256] bf16 (attn-proj out)
    u16b* __restrict__ hAll,           // [T*16][256] bf16
    unsigned* __restrict__ barCnt)
{
  const int T = 256, S = 256, D = 256;
  int blk = blockIdx.x;                // 0..15
  int tid = threadIdx.x, lane = tid & 31, w = tid >> 5;

  __shared__ float sw [256];           // softmax weights
  __shared__ float red[256];           // reductions / raw scores
  __shared__ float xc [320];           // [x_t, context] for this batch row
  __shared__ float cS [256];           // c for cols [16*blk,16*blk+16), [m*16+j]
  __shared__ float gbuf[4][256];       // i,f,g,o tiles, [m*16+j]

  unsigned bexp = 0;

  { // init: this block owns state cols [blk*16, blk*16+16)
    int m = tid >> 4, j = tid & 15, col = blk * 16 + j;
    float hv = h0[m * 256 + col];
    hF [m * 256 + col] = hv;
    hbf[m * 256 + col] = f2bf(hv);
    cS [tid] = c0[m * 256 + col];
  }
  gbar(barCnt, bexp += 16);

  for (int t = 0; t < T; ++t) {
    int b = blk;
    // ---- scores[s] = h[b] . enc[b][s],  s = tid --------------------------
    {
      const float* hb = hF + b * D;
      const float* er = encF + ((size_t)(b * S) + tid) * D;
      float acc = 0.f;
      for (int d = 0; d < D; ++d) acc += hb[d] * er[d];
      red[tid] = acc;
      sw[tid]  = acc;
    }
    __syncthreads();
    // ---- block softmax over 256 values -----------------------------------
    for (int off = 128; off > 0; off >>= 1) {
      if (tid < off) sw[tid] = fmaxf(sw[tid], sw[tid + off]);
      __syncthreads();
    }
    float mx = sw[0];
    __syncthreads();
    float e = expf(red[tid] - mx);
    sw[tid] = e;
    red[tid] = e;
    __syncthreads();
    for (int off = 128; off > 0; off >>= 1) {
      if (tid < off) red[tid] += red[tid + off];
      __syncthreads();
    }
    float inv = 1.f / red[0];
    __syncthreads();
    sw[tid] *= inv;
    // ---- context[d] = sum_s w[s] enc[b][s][d],  d = tid; build xcat ------
    if (tid < 64) xc[tid] = tgtE[((size_t)t * 16 + b) * 64 + tid];
    __syncthreads();
    {
      float acc = 0.f;
      for (int s = 0; s < S; ++s) acc += sw[s] * encF[((size_t)(b * S) + s) * D + tid];
      xc[64 + tid] = acc;
    }
    __syncthreads();
    // ---- attn proj: x[b][n] = attnB[n] + xc . attnW[n],  n = tid ----------
    {
      const float* wr = attnW + (size_t)tid * 320;
      float acc = attnB[tid];
      for (int k = 0; k < 320; ++k) acc += xc[k] * wr[k];
      xbf[b * 256 + tid] = f2bf(acc);
    }
    gbar(barCnt, bexp += 16);

    // ---- LSTM gates (block = col-tile blk): wave g computes gate g --------
    if (w < 4) {
      int n0 = (w * 16 + blk) * 16;                 // gate w, col-tile blk
      v8f acc = bias_splat(decB, n0, lane);
#pragma unroll
      for (int kk = 0; kk < 8; ++kk)
        acc = wmma_bf16(load_a(xbf, 256, kk * 32, lane),
                        load_b(decWih, 256, n0, kk * 32, lane), acc);
#pragma unroll
      for (int kk = 0; kk < 8; ++kk)
        acc = wmma_bf16(load_a(hbf, 256, kk * 32, lane),
                        load_b(decWhh, 256, n0, kk * 32, lane), acc);
      int j = lane & 15, mbb = (lane >> 4) << 3;
#pragma unroll
      for (int r = 0; r < 8; ++r) gbuf[w][(mbb + r) * 16 + j] = acc[r];
    }
    __syncthreads();
    // ---- cell update for this block's 16x16 col slice ---------------------
    {
      int m = tid >> 4, j = tid & 15, col = blk * 16 + j;
      float iv = sigmoidf_(gbuf[0][tid]);
      float fv = sigmoidf_(gbuf[1][tid]);
      float gv = tanhf(gbuf[2][tid]);
      float ov = sigmoidf_(gbuf[3][tid]);
      float cN = fv * cS[tid] + iv * gv;
      float hN = ov * tanhf(cN);
      cS[tid] = cN;
      hF[m * 256 + col] = hN;
      u16b hb = f2bf(hN);
      hbf[m * 256 + col] = hb;
      hAll[((size_t)t * 16 + m) * 256 + col] = hb;
    }
    gbar(barCnt, bexp += 16);
  }
}

// ---------------------------------------------------------------------------
// fc GEMM: out[b*256+t, v] = (hAll @ fcW.T + fcB)[t*16+b, v].  M=4096, N=32000, K=256.
// Grid (125, 8), block 8 waves. Wave holds B frags for 2 N-tiles in registers
// (128 VGPRs) and loops 32 M-tiles; the shared 8KB A tile is async-DMA'd into
// LDS with double buffering (global_load_async_to_lds_b128 / ASYNCcnt).
// ---------------------------------------------------------------------------
__global__ __launch_bounds__(256) void k_fc(const u16b* __restrict__ hAll,
                                            const u16b* __restrict__ fcW,
                                            const float* __restrict__ fcB,
                                            float* __restrict__ out) {
  __shared__ u16b aT[2][16 * 256];     // 8KB double-buffered A tile

  int tid = threadIdx.x, lane = tid & 31, w = tid >> 5;
  int n0a = blockIdx.x * 256 + w * 32; // wave's two N-tiles: n0a, n0a+16
  int mt0 = blockIdx.y * 32;           // 32 consecutive M-tiles (t index)

  v16bf bfr[2][8];                     // register-resident B fragments
#pragma unroll
  for (int p = 0; p < 2; ++p)
#pragma unroll
    for (int kk = 0; kk < 8; ++kk)
      bfr[p][kk] = load_b(fcW, 256, n0a + p * 16, kk * 32, lane);
  v8f bia[2];
#pragma unroll
  for (int p = 0; p < 2; ++p) bia[p] = bias_splat(fcB, n0a + p * 16, lane);

  stage_a(aT[0], hAll + (size_t)mt0 * 4096, tid);

  for (int mi = 0; mi < 32; ++mi) {
    int cur = mi & 1;
    wait_async0();                     // current A tile landed in LDS
    __syncthreads();                   // ... and prior readers of aT[cur^1] done
    if (mi + 1 < 32) stage_a(aT[cur ^ 1], hAll + (size_t)(mt0 + mi + 1) * 4096, tid);

    v16bf a[8];
#pragma unroll
    for (int kk = 0; kk < 8; ++kk) a[kk] = load_a(aT[cur], 256, kk * 32, lane);

    int tt = mt0 + mi;
#pragma unroll
    for (int p = 0; p < 2; ++p) {
      v8f acc = bia[p];
#pragma unroll
      for (int kk = 0; kk < 8; ++kk) acc = wmma_bf16(a[kk], bfr[p][kk], acc);
      int col = n0a + p * 16 + (lane & 15);
      int mbb = (lane >> 4) << 3;
#pragma unroll
      for (int r = 0; r < 8; ++r) {
        int b = mbb + r;
        out[((size_t)b * 256 + tt) * 32000 + col] = acc[r];   // out[b][t][v]
      }
    }
  }
}

// ---------------------------------------------------------------------------
extern "C" void kernel_launch(void* const* d_in, const int* in_sizes, int n_in,
                              void* d_out, int out_size, void* d_ws, size_t ws_size,
                              hipStream_t stream) {
  const int*   encrypt = (const int*)d_in[0];
  const int*   decrypt = (const int*)d_in[1];
  const float* emb     = (const float*)d_in[2];
  const float* efWih   = (const float*)d_in[3];
  const float* efWhh   = (const float*)d_in[4];
  const float* efB     = (const float*)d_in[5];
  const float* ebWih   = (const float*)d_in[6];
  const float* ebWhh   = (const float*)d_in[7];
  const float* ebB     = (const float*)d_in[8];
  const float* dWih    = (const float*)d_in[9];
  const float* dWhh    = (const float*)d_in[10];
  const float* dB      = (const float*)d_in[11];
  const float* aW      = (const float*)d_in[12];
  const float* aB      = (const float*)d_in[13];
  const float* fW      = (const float*)d_in[14];
  const float* fB      = (const float*)d_in[15];
  float* out = (float*)d_out;

  char* ws = (char*)d_ws;
  size_t off = 0;
  auto alloc = [&](size_t bytes) -> void* {
    void* p = ws + off;
    off = (off + bytes + 255) & ~(size_t)255;
    return p;
  };

  u16b*  wEfIh = (u16b*)alloc((size_t)512 * 64 * 2);
  u16b*  wEfHh = (u16b*)alloc((size_t)512 * 128 * 2);
  u16b*  wEbIh = (u16b*)alloc((size_t)512 * 64 * 2);
  u16b*  wEbHh = (u16b*)alloc((size_t)512 * 128 * 2);
  u16b*  wDIh  = (u16b*)alloc((size_t)1024 * 256 * 2);
  u16b*  wDHh  = (u16b*)alloc((size_t)1024 * 256 * 2);
  u16b*  wF    = (u16b*)alloc((size_t)32000 * 256 * 2);
  u16b*  srcE  = (u16b*)alloc((size_t)256 * 16 * 64 * 2);
  float* tgtEf = (float*)alloc((size_t)256 * 16 * 64 * 4);
  float* encF  = (float*)alloc((size_t)16 * 256 * 256 * 4);
  float* h0    = (float*)alloc((size_t)16 * 256 * 4);
  float* c0    = (float*)alloc((size_t)16 * 256 * 4);
  u16b*  hAll  = (u16b*)alloc((size_t)256 * 16 * 256 * 2);
  float* hF    = (float*)alloc((size_t)16 * 256 * 4);
  u16b*  hbf   = (u16b*)alloc((size_t)16 * 256 * 2);
  u16b*  xbf   = (u16b*)alloc((size_t)16 * 256 * 2);
  unsigned* barCnt = (unsigned*)alloc(256);

  // Phase 0: weight conversion + barrier reset
  k_zero<<<1, 256, 0, stream>>>(barCnt);
  k_f32_to_bf16<<<128,  256, 0, stream>>>(efWih, wEfIh, 512 * 64);
  k_f32_to_bf16<<<256,  256, 0, stream>>>(efWhh, wEfHh, 512 * 128);
  k_f32_to_bf16<<<128,  256, 0, stream>>>(ebWih, wEbIh, 512 * 64);
  k_f32_to_bf16<<<256,  256, 0, stream>>>(ebWhh, wEbHh, 512 * 128);
  k_f32_to_bf16<<<512,  256, 0, stream>>>(dWih,  wDIh,  1024 * 256);
  k_f32_to_bf16<<<512,  256, 0, stream>>>(dWhh,  wDHh,  1024 * 256);
  k_f32_to_bf16<<<2048, 256, 0, stream>>>(fW,    wF,    32000 * 256);

  // Phase 1: embeddings
  k_embed_bf <<<512, 256, 0, stream>>>(encrypt, emb, srcE,  256);
  k_embed_f32<<<512, 256, 0, stream>>>(decrypt, emb, tgtEf, 256);

  // Phase 2: encoder (block 0 = fwd, block 1 = bwd)
  k_encoder<<<2, 256, 0, stream>>>(srcE, wEfIh, wEfHh, efB, wEbIh, wEbHh, ebB,
                                   encF, h0, c0);

  // Phase 3: decoder recurrence (16 cooperative blocks)
  k_decoder<<<16, 256, 0, stream>>>(tgtEf, encF, h0, c0, aW, aB,
                                    wDIh, wDHh, dB, hF, hbf, xbf, hAll, barCnt);

  // Phase 4: fc GEMM (67 GFLOP bulk; output-write bound)
  k_fc<<<dim3(125, 8), 256, 0, stream>>>(hAll, wF, fB, out);

  (void)in_sizes; (void)n_in; (void)out_size; (void)ws_size;
}
